// MultiheadAttention_83975200571962
// MI455X (gfx1250) — matrix operand order repaired and checked by reading, not compiled, not run on hardware
//
#include <hip/hip_runtime.h>
#include <hip/hip_bf16.h>

// ---- problem constants (from reference) ----
#define B_SZ   16
#define S_LEN  4096
#define E_DIM  2048
#define H_NUM  32
#define D_HEAD 64

typedef __attribute__((ext_vector_type(16))) __bf16 v16bf;
typedef __attribute__((ext_vector_type(8)))  __bf16 v8bf;
typedef __attribute__((ext_vector_type(8)))  float  v8f;
typedef __attribute__((ext_vector_type(4)))  float  f4;

__device__ __forceinline__ float wave_sum(float v) {
  #pragma unroll
  for (int off = 16; off > 0; off >>= 1) v += __shfl_xor(v, off, 32);
  return v;
}
__device__ __forceinline__ float wave_max(float v) {
  #pragma unroll
  for (int off = 16; off > 0; off >>= 1) v = fmaxf(v, __shfl_xor(v, off, 32));
  return v;
}
__device__ __forceinline__ float dot4(const f4 a, const f4 b) {
  return a.x * b.x + a.y * b.y + a.z * b.z + a.w * b.w;
}
__device__ __forceinline__ v16bf cat8(const v8bf lo, const v8bf hi) {
  return __builtin_shufflevector(lo, hi, 0,1,2,3,4,5,6,7,8,9,10,11,12,13,14,15);
}

// ---------------------------------------------------------------------------
// Kernel A: q[b,:] = query[b,S-1,:] @ Wq^T (softmax scale folded in), then
//           qk[b,h,e] = sum_j q[b,h*64+j] * Wk[h*64+j,e]  -> bf16
// grid (B, H), 256 threads
// ---------------------------------------------------------------------------
__global__ __launch_bounds__(256) void mha_proj_qk(
    const float* __restrict__ query, const float* __restrict__ Win,
    __bf16* __restrict__ qk)
{
  const int b = blockIdx.x, h = blockIdx.y;
  const int t = threadIdx.x, w = t >> 5, lane = t & 31;
  __shared__ float q_h[D_HEAD];

  const float* qlast = query + ((size_t)b * S_LEN + (S_LEN - 1)) * E_DIM;
  const float* Wq = Win;                                 // rows [0, E)
  const float* Wk = Win + (size_t)E_DIM * E_DIM;         // rows [E, 2E)
  const float scale = 0.125f;                            // 1/sqrt(64)

  #pragma unroll
  for (int jj = 0; jj < 8; ++jj) {
    const int j = w * 8 + jj;
    const float* wr = Wq + (size_t)(h * D_HEAD + j) * E_DIM;
    float s = 0.f;
    for (int e = lane * 4; e < E_DIM; e += 128)
      s += dot4(*(const f4*)(qlast + e), *(const f4*)(wr + e));
    s = wave_sum(s);
    if (lane == 0) q_h[j] = s * scale;
  }
  __syncthreads();

  for (int e = t; e < E_DIM; e += 256) {
    float acc = 0.f;
    #pragma unroll 8
    for (int j = 0; j < D_HEAD; ++j)
      acc += q_h[j] * Wk[(size_t)(h * D_HEAD + j) * E_DIM + e];
    qk[((size_t)(b * H_NUM + h)) * E_DIM + e] = (__bf16)acc;
  }
}

// ---------------------------------------------------------------------------
// Stage helper: thread t owns 16 contiguous floats of one tile row.
//   row = t>>2, col = (t&3)*16  (256 threads -> 64 rows x 64 cols)
// 4x global_load_b128 -> packed bf16 -> 2x ds_store_b128.
// ---------------------------------------------------------------------------
__device__ __forceinline__ void stage16(const float* __restrict__ src,
                                        __bf16* __restrict__ dst) {
  f4 x0 = *(const f4*)(src);
  f4 x1 = *(const f4*)(src + 4);
  f4 x2 = *(const f4*)(src + 8);
  f4 x3 = *(const f4*)(src + 12);
  v8bf lo, hi;
  #pragma unroll
  for (int i = 0; i < 4; ++i) {
    lo[i]     = (__bf16)x0[i]; lo[4 + i] = (__bf16)x1[i];
    hi[i]     = (__bf16)x2[i]; hi[4 + i] = (__bf16)x3[i];
  }
  *(v8bf*)(dst)     = lo;
  *(v8bf*)(dst + 8) = hi;
}

#define LDS_PAD 8          // row stride 72 elems = 144 B (16B-aligned)

// ---------------------------------------------------------------------------
// Kernel B: scores[b,h,s] = query[b,s,:] . qk[b,h,:]   (bf16 WMMA, f32 acc)
// GEMM per b: M=S(4096) x N=H(32) x K=E(2048), K-step 64 (2 WMMA per stage).
// Block = 256 thr (8 waves) covers 64 s-rows x 32 h-cols. grid (S/64, B)
// ---------------------------------------------------------------------------
__global__ __launch_bounds__(256) void mha_scores_wmma(
    const float* __restrict__ query, const __bf16* __restrict__ qk,
    float* __restrict__ scores)
{
  const int b  = blockIdx.y;
  const int s0 = blockIdx.x * 64;
  const int t = threadIdx.x, lane = t & 31, w = t >> 5;
  const int wsub = w >> 1;          // s sub-tile 0..3
  const int wh   = w & 1;           // h tile 0..1
  const int half = lane >> 4, l15 = lane & 15;

  __shared__ __bf16 lds_q[64][64 + LDS_PAD];   // 64 s x 64 k

  const int srow = t >> 2, scol = (t & 3) * 16;
  const float* qbase = query + (size_t)b * S_LEN * E_DIM;
  const float* sptr  = qbase + (size_t)(s0 + srow) * E_DIM + scol;
  const int m = wsub * 16 + l15;
  const __bf16* qkrow = qk + ((size_t)(b * H_NUM + wh * 16 + l15)) * E_DIM + half * 16;

  v8f acc = {};

  for (int ke = 0; ke < E_DIM; ke += 64) {
    stage16(sptr + ke, &lds_q[srow][scol]);
    if (ke + 64 < E_DIM) __builtin_prefetch(sptr + ke + 64, 0, 3);
    __syncthreads();

    #pragma unroll
    for (int ks2 = 0; ks2 < 2; ++ks2) {
      const int koff = ks2 * 32;
      // A fragment: 16x32 (M=s, K=e); lane m, half 0: K 0..15, half 1: K 16..31
      v8bf alo = *(const v8bf*)&lds_q[m][koff + half * 16];
      v8bf ahi = *(const v8bf*)&lds_q[m][koff + half * 16 + 8];
      // B fragment: 32x16 (K=e, N=h); lane n=l15, half 0: K 0..15, half 1: K 16..31
      v16bf bfrag = *(const v16bf*)(qkrow + ke + koff);
      acc = __builtin_amdgcn_wmma_f32_16x16x32_bf16(
          false, cat8(alo, ahi), false, bfrag, (short)0, acc, false, false);
    }
    __syncthreads();
  }

  // C tile: lane n=l15 -> h, VGPR i -> m = half*8+i -> s (8 consecutive s)
  const int sbase = s0 + wsub * 16 + half * 8;
  const int hcol  = wh * 16 + l15;
  float* drow = scores + ((size_t)(b * H_NUM + hcol)) * S_LEN + sbase;
  f4 o0 = {acc[0], acc[1], acc[2], acc[3]};
  f4 o1 = {acc[4], acc[5], acc[6], acc[7]};
  *(f4*)(drow)     = o0;
  *(f4*)(drow + 4) = o1;
}

// ---------------------------------------------------------------------------
// Kernel C: row softmax over S=4096, output bf16 probs. grid (B*H), 256 thr
// ---------------------------------------------------------------------------
__global__ __launch_bounds__(256) void mha_softmax(
    const float* __restrict__ scores, __bf16* __restrict__ probs)
{
  const int row = blockIdx.x;                 // b*H + h
  const float* src = scores + (size_t)row * S_LEN;
  __bf16* dst = probs + (size_t)row * S_LEN;
  const int t = threadIdx.x, lane = t & 31, w = t >> 5;
  __shared__ float red[8];

  float vals[16];
  float m = -INFINITY;
  #pragma unroll
  for (int i = 0; i < 16; ++i) { vals[i] = src[t + i * 256]; m = fmaxf(m, vals[i]); }
  m = wave_max(m);
  if (lane == 0) red[w] = m;
  __syncthreads();
  float gm = red[0];
  #pragma unroll
  for (int i = 1; i < 8; ++i) gm = fmaxf(gm, red[i]);
  __syncthreads();

  float l = 0.f;
  #pragma unroll
  for (int i = 0; i < 16; ++i) { vals[i] = __expf(vals[i] - gm); l += vals[i]; }
  l = wave_sum(l);
  if (lane == 0) red[w] = l;
  __syncthreads();
  float gl = 0.f;
  #pragma unroll
  for (int i = 0; i < 8; ++i) gl += red[i];
  const float inv = 1.f / gl;
  #pragma unroll
  for (int i = 0; i < 16; ++i) dst[t + i * 256] = (__bf16)(vals[i] * inv);
}

// ---------------------------------------------------------------------------
// Kernel D: r[b,h,e] = sum_s probs[b,h,s] * query[b,s,e]  (bf16 WMMA, f32 acc)
// GEMM per b: M=H(32) x N=E(2048) x K=S(4096), K-step 64.
// Block = 256 thr covers 32 h-rows x 64 e-cols; grid (E/64, B)
// ---------------------------------------------------------------------------
__global__ __launch_bounds__(256) void mha_ctx_wmma(
    const float* __restrict__ query, const __bf16* __restrict__ probs,
    float* __restrict__ r)
{
  const int b  = blockIdx.y;
  const int n0 = blockIdx.x * 64;
  const int t = threadIdx.x, lane = t & 31, w = t >> 5;
  const int wm = w & 1;            // h tile 0..1
  const int wn = w >> 1;           // e sub-tile 0..3
  const int half = lane >> 4, l15 = lane & 15;

  __shared__ __bf16 lds_b[64][64 + LDS_PAD];   // 64 k(s) x 64 n(e)

  const int krow = t >> 2, kcol = (t & 3) * 16;
  const float* qbase = query + (size_t)b * S_LEN * E_DIM + n0 + kcol;
  const __bf16* prow = probs + ((size_t)(b * H_NUM + wm * 16 + l15)) * S_LEN;

  v8f acc = {};

  for (int ks = 0; ks < S_LEN; ks += 64) {
    stage16(qbase + (size_t)(ks + krow) * E_DIM, &lds_b[krow][kcol]);
    if (ks + 64 < S_LEN) __builtin_prefetch(qbase + (size_t)(ks + 64 + krow) * E_DIM, 0, 3);
    __syncthreads();

    #pragma unroll
    for (int ks2 = 0; ks2 < 2; ++ks2) {
      const int koff = ks2 * 32;
      // A fragment: probs 16x32 (M=h, K=s), half 0: K 0..15, half 1: K 16..31
      v8bf alo = *(const v8bf*)(prow + ks + koff + half * 16);
      v8bf ahi = *(const v8bf*)(prow + ks + koff + half * 16 + 8);
      // B fragment: 32x16 (K=s, N=e) column gather from LDS
      v16bf bfrag;
      #pragma unroll
      for (int i = 0; i < 16; ++i) bfrag[i] = lds_b[koff + half * 16 + i][wn * 16 + l15];
      acc = __builtin_amdgcn_wmma_f32_16x16x32_bf16(
          false, cat8(alo, ahi), false, bfrag, (short)0, acc, false, false);
    }
    __syncthreads();
  }

  // C tile: n=l15 -> e, VGPR i -> m = half*8+i -> h
  #pragma unroll
  for (int i = 0; i < 8; ++i) {
    const int hrow = wm * 16 + half * 8 + i;
    const int ecol = n0 + wn * 16 + l15;
    r[((size_t)(b * H_NUM + hrow)) * E_DIM + ecol] = acc[i];
  }
}

// ---------------------------------------------------------------------------
// Kernel E1: ctx[b, h*64+j] = r[b,h,:] . Wv[h*64+j,:].  grid (B, H)
// ---------------------------------------------------------------------------
__global__ __launch_bounds__(256) void mha_vproj(
    const float* __restrict__ r, const float* __restrict__ Win,
    float* __restrict__ ctx)
{
  const int b = blockIdx.x, h = blockIdx.y;
  const int t = threadIdx.x, w = t >> 5, lane = t & 31;
  const float* Wv = Win + 2 * (size_t)E_DIM * E_DIM;     // rows [2E, 3E)
  const float* rrow = r + ((size_t)(b * H_NUM + h)) * E_DIM;
  #pragma unroll
  for (int jj = 0; jj < 8; ++jj) {
    const int j = w * 8 + jj;
    const float* wr = Wv + (size_t)(h * D_HEAD + j) * E_DIM;
    float s = 0.f;
    for (int e = lane * 4; e < E_DIM; e += 128)
      s += dot4(*(const f4*)(rrow + e), *(const f4*)(wr + e));
    s = wave_sum(s);
    if (lane == 0) ctx[(size_t)b * E_DIM + h * D_HEAD + j] = s;
  }
}

// ---------------------------------------------------------------------------
// Kernel E2: out[b, eo] = ctx[b,:] . Wo[eo,:].  grid (B, E/64)
// ---------------------------------------------------------------------------
__global__ __launch_bounds__(256) void mha_oproj(
    const float* __restrict__ ctx, const float* __restrict__ Wo,
    float* __restrict__ out)
{
  const int b = blockIdx.x, g = blockIdx.y;
  const int t = threadIdx.x, w = t >> 5, lane = t & 31;
  const float* crow = ctx + (size_t)b * E_DIM;
  #pragma unroll
  for (int jj = 0; jj < 8; ++jj) {
    const int eo = g * 64 + w * 8 + jj;
    const float* wr = Wo + (size_t)eo * E_DIM;
    float s = 0.f;
    for (int e = lane * 4; e < E_DIM; e += 128)
      s += dot4(*(const f4*)(crow + e), *(const f4*)(wr + e));
    s = wave_sum(s);
    if (lane == 0) out[(size_t)b * E_DIM + eo] = s;
  }
}

// ---------------------------------------------------------------------------
extern "C" void kernel_launch(void* const* d_in, const int* in_sizes, int n_in,
                              void* d_out, int out_size, void* d_ws, size_t ws_size,
                              hipStream_t stream) {
  (void)in_sizes; (void)n_in; (void)out_size; (void)ws_size;
  const float* query = (const float*)d_in[0];   // (16, 4096, 2048) f32
  const float* Win   = (const float*)d_in[1];   // (6144, 2048) f32
  const float* Wo    = (const float*)d_in[2];   // (2048, 2048) f32
  float* out = (float*)d_out;                   // (16, 1, 2048) f32

  char* ws = (char*)d_ws;
  __bf16* qk     = (__bf16*)(ws);                          //  2 MB  B*H*E bf16
  float*  scores = (float*) (ws + ( 2u << 20));            //  8 MB  B*H*S f32
  __bf16* probs  = (__bf16*)(ws + (10u << 20));            //  4 MB  B*H*S bf16
  float*  r      = (float*) (ws + (14u << 20));            //  4 MB  B*H*E f32
  float*  ctx    = (float*) (ws + (18u << 20));            // 128 KB B*E   f32

  mha_proj_qk    <<<dim3(B_SZ, H_NUM),       256, 0, stream>>>(query, Win, qk);
  mha_scores_wmma<<<dim3(S_LEN / 64, B_SZ),  256, 0, stream>>>(query, qk, scores);
  mha_softmax    <<<dim3(B_SZ * H_NUM),      256, 0, stream>>>(scores, probs);
  mha_ctx_wmma   <<<dim3(E_DIM / 64, B_SZ),  256, 0, stream>>>(query, probs, r);
  mha_vproj      <<<dim3(B_SZ, H_NUM),       256, 0, stream>>>(r, Win, ctx);
  mha_oproj      <<<dim3(B_SZ, E_DIM / 64),  256, 0, stream>>>(ctx, Wo, out);
}
